// Select_14250701488402
// MI455X (gfx1250) — compile-verified, matrix operand order checked by reading, not run on hardware
//
#include <hip/hip_runtime.h>

#define TOPK 5
#define BB   32
#define SX   512
#define SM   4096
#define DD   256

typedef _Float16 v8h  __attribute__((ext_vector_type(8)));
typedef _Float16 v16h __attribute__((ext_vector_type(16)));
typedef float    v8f  __attribute__((ext_vector_type(8)));
typedef float    v4f  __attribute__((ext_vector_type(4)));

#define NEG_INF (-3.402823466e+38f)

// ---------------------------------------------------------------------------
// Kernel 1: per-row L2 norm of x, write f16(x / ||x||).  One wave per row
// (D=256 -> 8 floats per lane), full-wave xor-shuffle reduction (wave32).
// ---------------------------------------------------------------------------
__global__ __launch_bounds__(256) void prep_x_kernel(const float* __restrict__ x,
                                                     _Float16* __restrict__ xh) {
  const int wid  = (int)((blockIdx.x * blockDim.x + threadIdx.x) >> 5);
  const int lane = (int)(threadIdx.x & 31);
  if (wid >= BB * SX) return;
  const float* p = x + (size_t)wid * DD + lane * 8;
  v4f a = *(const v4f*)p;
  v4f b = *(const v4f*)(p + 4);
  float s = a[0]*a[0]+a[1]*a[1]+a[2]*a[2]+a[3]*a[3]
          + b[0]*b[0]+b[1]*b[1]+b[2]*b[2]+b[3]*b[3];
  #pragma unroll
  for (int off = 16; off; off >>= 1) s += __shfl_xor(s, off, 32);
  const float rs = 1.0f / sqrtf(s);
  v8h o;
  o[0] = (_Float16)(a[0]*rs); o[1] = (_Float16)(a[1]*rs);
  o[2] = (_Float16)(a[2]*rs); o[3] = (_Float16)(a[3]*rs);
  o[4] = (_Float16)(b[0]*rs); o[5] = (_Float16)(b[1]*rs);
  o[6] = (_Float16)(b[2]*rs); o[7] = (_Float16)(b[3]*rs);
  *(v8h*)(xh + (size_t)wid * DD + lane * 8) = o;
}

// ---------------------------------------------------------------------------
// Kernel 2: f16 copy of memory rows (unnormalized) + 1/||m|| per row.
// ---------------------------------------------------------------------------
__global__ __launch_bounds__(256) void prep_mem_kernel(const float* __restrict__ mem,
                                                       _Float16* __restrict__ mh,
                                                       float* __restrict__ invm) {
  const int wid  = (int)((blockIdx.x * blockDim.x + threadIdx.x) >> 5);
  const int lane = (int)(threadIdx.x & 31);
  if (wid >= BB * SM) return;
  const float* p = mem + (size_t)wid * DD + lane * 8;
  v4f a = *(const v4f*)p;
  v4f b = *(const v4f*)(p + 4);
  float s = a[0]*a[0]+a[1]*a[1]+a[2]*a[2]+a[3]*a[3]
          + b[0]*b[0]+b[1]*b[1]+b[2]*b[2]+b[3]*b[3];
  #pragma unroll
  for (int off = 16; off; off >>= 1) s += __shfl_xor(s, off, 32);
  v8h o;
  o[0] = (_Float16)a[0]; o[1] = (_Float16)a[1];
  o[2] = (_Float16)a[2]; o[3] = (_Float16)a[3];
  o[4] = (_Float16)b[0]; o[5] = (_Float16)b[1];
  o[6] = (_Float16)b[2]; o[7] = (_Float16)b[3];
  *(v8h*)(mh + (size_t)wid * DD + lane * 8) = o;
  if (lane == 0) invm[wid] = 1.0f / sqrtf(s);
}

// ---------------------------------------------------------------------------
// Kernel 3: WMMA GEMM + row-max.  One wave owns one 16-row M-tile; the full
// 16x256 A tile lives in 8 v16h fragments (64 VGPRs).  Sweep 32 N-tiles x
// 8 K-steps of v_wmma_f32_16x16x32_f16, track per-row running max, reduce
// across the 16-lane halves, scale by 1/||m||, store sim_max[b][m].
// Grid is sized so every wave is fully populated (EXEC all 1s for WMMA).
// ---------------------------------------------------------------------------
__global__ __launch_bounds__(256) void simmax_kernel(const _Float16* __restrict__ mh,
                                                     const _Float16* __restrict__ xh,
                                                     const float* __restrict__ invm,
                                                     float* __restrict__ smax) {
  const int wid  = (int)((blockIdx.x << 3) + (threadIdx.x >> 5)); // 8192 waves
  const int lane = (int)(threadIdx.x & 31);
  const int b    = wid >> 8;            // 256 M-tiles per batch
  const int m0   = (wid & 255) << 4;    // first memory row of this tile
  const int r    = lane & 15;           // row within tile (M for A, N for B)
  const int hb   = (lane >> 4) & 1;     // K-half select per ISA 16-bit layout

  // A (memory) fragments: lane<16 holds K 0-7 / 16-23, lane>=16 holds 8-15 / 24-31.
  const char* abase = (const char*)(mh + ((size_t)b * SM + m0 + r) * DD) + hb * 16;
  v16h afrag[8];
  #pragma unroll
  for (int kk = 0; kk < 8; ++kk) {
    union { v16h v; v8h h[2]; } u;
    u.h[0] = *(const v8h*)(abase + kk * 64);
    u.h[1] = *(const v8h*)(abase + kk * 64 + 32);
    afrag[kk] = u.v;
  }

  v8f rowmax;
  #pragma unroll
  for (int i = 0; i < 8; ++i) rowmax[i] = NEG_INF;

  const char* bcol = (const char*)(xh + ((size_t)b * SX + r) * DD) + hb * 16;
  for (int nt = 0; nt < 32; ++nt) {                 // 32 N-tiles cover SX=512
    const char* bbase = bcol + (size_t)nt * 16 * DD * 2;
    v8f acc = {};
    #pragma unroll
    for (int kk = 0; kk < 8; ++kk) {                // K = 256 in steps of 32
      union { v16h v; v8h h[2]; } u;
      u.h[0] = *(const v8h*)(bbase + kk * 64);
      u.h[1] = *(const v8h*)(bbase + kk * 64 + 32);
      acc = __builtin_amdgcn_wmma_f32_16x16x32_f16(false, afrag[kk], false, u.v,
                                                   (short)0, acc, false, false);
    }
    #pragma unroll
    for (int i = 0; i < 8; ++i) rowmax[i] = fmaxf(rowmax[i], acc[i]);
  }

  // Max across the 16 lanes of each half (N direction); xor masks stay in-half.
  #pragma unroll
  for (int off = 8; off; off >>= 1) {
    #pragma unroll
    for (int i = 0; i < 8; ++i)
      rowmax[i] = fmaxf(rowmax[i], __shfl_xor(rowmax[i], off, 32));
  }

  // C/D layout: lanes 0-15 hold M = reg (0..7), lanes 16-31 hold M = reg+8.
  if (r == 0) {
    const int mb = m0 + (hb ? 8 : 0);
    #pragma unroll
    for (int i = 0; i < 8; ++i) {
      const size_t idx = (size_t)b * SM + mb + i;
      smax[idx] = rowmax[i] * invm[idx];
    }
  }
}

// ---------------------------------------------------------------------------
// Kernel 4: per-batch top-5 (5 masked argmax passes, ties -> lower index to
// match jax.lax.top_k), gather original f32 rows, emit indices as floats.
// ---------------------------------------------------------------------------
__global__ __launch_bounds__(256) void topk_gather_kernel(const float* __restrict__ smax,
                                                          const float* __restrict__ mem,
                                                          float* __restrict__ out) {
  const int b = (int)blockIdx.x;
  const int t = (int)threadIdx.x;
  __shared__ float sv[256];
  __shared__ int   si[256];
  __shared__ int   sel[TOPK];
  const float* row = smax + (size_t)b * SM;

  for (int k = 0; k < TOPK; ++k) {
    float bv = NEG_INF;
    int   bi = SM;
    for (int m = t; m < SM; m += 256) {
      bool taken = false;
      for (int j = 0; j < k; ++j) taken |= (sel[j] == m);
      if (taken) continue;
      const float v = row[m];
      if (v > bv || (v == bv && m < bi)) { bv = v; bi = m; }
    }
    sv[t] = bv; si[t] = bi;
    __syncthreads();
    for (int s = 128; s; s >>= 1) {
      if (t < s) {
        const float ov = sv[t + s]; const int oi = si[t + s];
        if (ov > sv[t] || (ov == sv[t] && oi < si[t])) { sv[t] = ov; si[t] = oi; }
      }
      __syncthreads();
    }
    if (t == 0) sel[k] = si[0];
    __syncthreads();
  }

  // Outputs flat: querys[:,0,:] ... querys[:,4,:] (each [B,D] f32), then indices.
  for (int k = 0; k < TOPK; ++k) {
    const int idx = sel[k];
    out[(size_t)k * (BB * DD) + (size_t)b * DD + t] =
        mem[((size_t)b * SM + idx) * DD + t];
  }
  if (t < TOPK) out[(size_t)TOPK * BB * DD + b * TOPK + t] = (float)sel[t];
}

// ---------------------------------------------------------------------------
extern "C" void kernel_launch(void* const* d_in, const int* in_sizes, int n_in,
                              void* d_out, int out_size, void* d_ws, size_t ws_size,
                              hipStream_t stream) {
  const float* x   = (const float*)d_in[0];  // [B, SX, D]
  const float* mem = (const float*)d_in[1];  // [B, SM, D]
  float* out = (float*)d_out;

  char* ws = (char*)d_ws;
  const size_t xhB = (size_t)BB * SX * DD * sizeof(_Float16); // 8 MB
  const size_t mhB = (size_t)BB * SM * DD * sizeof(_Float16); // 64 MB
  _Float16* xh   = (_Float16*)ws;
  _Float16* mh   = (_Float16*)(ws + xhB);
  float*    invm = (float*)(ws + xhB + mhB);                  // 512 KB
  float*    smax = invm + (size_t)BB * SM;                    // 512 KB

  // wave-per-row; 8 waves per 256-thread block, grids divide exactly.
  prep_x_kernel  <<<(BB * SX) / 8, 256, 0, stream>>>(x, xh);
  prep_mem_kernel<<<(BB * SM) / 8, 256, 0, stream>>>(mem, mh, invm);
  // one wave per 16-row tile: 32 * 256 = 8192 waves -> 1024 blocks of 8 waves.
  simmax_kernel  <<<(BB * (SM / 16)) / 8, 256, 0, stream>>>(mh, xh, invm, smax);
  topk_gather_kernel<<<BB, 256, 0, stream>>>(smax, mem, out);
}
